// RecurrentSlotEncoder_40261023432922
// MI455X (gfx1250) — compile-verified, matrix operand order checked by reading, not run on hardware
//
#include <hip/hip_runtime.h>

// ---------------------------------------------------------------------------
// RecurrentSlotEncoder on MI455X (gfx1250, wave32, WMMA).
//   Phase A (kv_proj): time-invariant K/V projections over all B*K*N rows,
//     69 GFLOP bf16-WMMA, HBM-roofline ~11.5us (268 MB H @ 23.3 TB/s).
//     K-pass computed operand-swapped (C[d][n] = Wk * H^T) so both K (row-
//     major [n][d]) and V (transposed [d][n]) epilogues are packed b128
//     stores; the bf16 H tiles are loaded once and reused as A (V) and B (K).
//   Phase B: 16 sequential fused step kernels (grid=B): Q / scores / softmax
//     / upd / GRU / FFN / LN, all GEMMs on v_wmma_f32_16x16x32_bf16, f32
//     epilogues. LDS rows padded +8 elements (stride % 256B == 32) to stagger
//     banks for the 16-row A-operand gathers.
// ---------------------------------------------------------------------------

typedef __attribute__((ext_vector_type(16))) __bf16 v16bf;
typedef __attribute__((ext_vector_type(8)))  __bf16 v8bf;
typedef __attribute__((ext_vector_type(8)))  float  v8f;

#define DEV __device__ __forceinline__

constexpr int Bb = 16, Kk = 16, Nn = 1024, Dd = 256, Ll = 8;
constexpr int LDP = Dd + 8;    // padded f32 row (D-sized buffers)
constexpr int SCP = Nn + 8;    // padded f32 row (scores)
constexpr int ATP = Nn + 16;   // padded bf16 row (attn)

DEV v16bf cat8(v8bf lo, v8bf hi) {
  v16bf r;
#pragma unroll
  for (int i = 0; i < 8; ++i) { r[i] = lo[i]; r[i + 8] = hi[i]; }
  return r;
}

// 16x32 bf16 operand gather from a row-major f32 buffer.
// Lane l is row/col (l&15); its 16 K-values are p[0..7] and p[16..23] where
// p = row + ks*32 + kb, kb = (l>>4)*8 (CDNA5 16-bit A/B lane layout).
DEV v16bf a_from_f32(const float* p) {
  v16bf a;
#pragma unroll
  for (int i = 0; i < 8; ++i) {
    a[i]     = (__bf16)p[i];
    a[i + 8] = (__bf16)p[16 + i];
  }
  return a;
}

// Same gather from row-major bf16: two 16B-aligned b128 loads.
DEV v16bf b_from_bf16(const __bf16* p) {
  v8bf lo = *(const v8bf*)p;
  v8bf hi = *(const v8bf*)(p + 16);
  return cat8(lo, hi);
}

DEV v8f wmma_bf(v16bf a, v16bf b, v8f c) {
  return __builtin_amdgcn_wmma_f32_16x16x32_bf16(false, a, false, b, (short)0,
                                                 c, false, false);
}

DEV float wave_max(float x) {
#pragma unroll
  for (int m = 16; m >= 1; m >>= 1) x = fmaxf(x, __shfl_xor(x, m, 32));
  return x;
}
DEV float wave_sum(float x) {
#pragma unroll
  for (int m = 16; m >= 1; m >>= 1) x += __shfl_xor(x, m, 32);
  return x;
}
DEV float sigm(float x) { return 1.f / (1.f + __expf(-x)); }
DEV float tanh_fast(float x) { return 1.f - 2.f / (1.f + __expf(2.f * x)); }

// ---------------------------------------------------------------------------
// Kernel 0: bf16 weight conversion + slots0 = mu + exp(log_sigma)*eps
// ---------------------------------------------------------------------------
__global__ void prep_kernel(const float* __restrict__ eps,
                            const float* __restrict__ mu,
                            const float* __restrict__ ls,
                            const float* __restrict__ Wq, const float* __restrict__ Wk,
                            const float* __restrict__ Wv, const float* __restrict__ Wih,
                            const float* __restrict__ Whh, const float* __restrict__ W1,
                            const float* __restrict__ W2,
                            __bf16* __restrict__ q, __bf16* __restrict__ kk,
                            __bf16* __restrict__ v, __bf16* __restrict__ ih,
                            __bf16* __restrict__ hh, __bf16* __restrict__ w1,
                            __bf16* __restrict__ w2, float* __restrict__ slots) {
  int i = blockIdx.x * blockDim.x + threadIdx.x;
  if (i < Dd * Dd) {
    q[i]  = (__bf16)Wq[i];
    kk[i] = (__bf16)Wk[i];
    v[i]  = (__bf16)Wv[i];
    w1[i] = (__bf16)W1[i];
    w2[i] = (__bf16)W2[i];
  }
  if (i < 3 * Dd * Dd) {
    ih[i] = (__bf16)Wih[i];
    hh[i] = (__bf16)Whh[i];
  }
  if (i < Bb * Ll * Dd) {
    int ld = i & (Ll * Dd - 1);
    slots[i] = mu[ld] + __expf(ls[ld]) * eps[i];
  }
}

// ---------------------------------------------------------------------------
// Kernel 1: K/V projections, bf16-WMMA, both epilogues packed b128 stores.
//   Kp[bk][n][d] row-major : K-pass operand-swapped (A = Wk, B = H tiles)
//   Vt[bk][d][n] transposed: V-pass normal      (A = H tiles, B = Wv)
// ---------------------------------------------------------------------------
__global__ __launch_bounds__(256, 1) void kv_proj_kernel(
    const float* __restrict__ H, const __bf16* __restrict__ WkB,
    const __bf16* __restrict__ WvB, const float* __restrict__ bk,
    const float* __restrict__ bv, __bf16* __restrict__ Kp,
    __bf16* __restrict__ Vt) {
  const int wg    = blockIdx.x;       // B*K*8 = 2048 workgroups
  const int chunk = wg & 7;           // 128-row chunk of N
  const int bkidx = wg >> 3;          // b*K + k
  const int wave  = threadIdx.x >> 5;
  const int lane  = threadIdx.x & 31;
  const int r     = lane & 15;
  const int kb    = (lane >> 4) * 8;  // K-offset within 32-block
  const int hi8   = (lane >> 4) * 8;  // row offset of this lane's C half
  const int nbase = chunk * 128 + wave * 16;

  const float* Hrow = H + ((size_t)bkidx * Nn + nbase + r) * Dd;

  // Warm L2 for the next 128-row chunk (speculative, DEV scope).
  __builtin_prefetch(Hrow + 128 * Dd, 0, 1);

  // Preload all 8 H tiles once (f32 -> bf16 operand layout); used as the
  // B-operand in the K pass and the A-operand in the V pass.
  v16bf Ht[8];
#pragma unroll
  for (int ks = 0; ks < 8; ++ks) Ht[ks] = a_from_f32(Hrow + ks * 32 + kb);

  // ---- K projection (swapped): C[d][n] = Wk[d][:] . H[n][:] ----
  // C lane: col = token (nbase + r), rows = 8 consecutive d -> packed store.
#pragma unroll 1
  for (int dt = 0; dt < 16; ++dt) {
    v8f c = {};
    const __bf16* arow = WkB + (size_t)(dt * 16 + r) * Dd;
#pragma unroll
    for (int ks = 0; ks < 8; ++ks)
      c = wmma_bf(b_from_bf16(arow + ks * 32 + kb), Ht[ks], c);
    const float4 bi0 = *(const float4*)&bk[dt * 16 + hi8];
    const float4 bi1 = *(const float4*)&bk[dt * 16 + hi8 + 4];
    v8bf pk;
    pk[0] = (__bf16)(c[0] + bi0.x); pk[1] = (__bf16)(c[1] + bi0.y);
    pk[2] = (__bf16)(c[2] + bi0.z); pk[3] = (__bf16)(c[3] + bi0.w);
    pk[4] = (__bf16)(c[4] + bi1.x); pk[5] = (__bf16)(c[5] + bi1.y);
    pk[6] = (__bf16)(c[6] + bi1.z); pk[7] = (__bf16)(c[7] + bi1.w);
    *(v8bf*)(Kp + ((size_t)bkidx * Nn + nbase + r) * Dd + dt * 16 + hi8) = pk;
  }

  // ---- V projection, stored transposed [d][n] (packed b128 store) ----
#pragma unroll 1
  for (int nt = 0; nt < 16; ++nt) {
    v8f c = {};
    const __bf16* wrow = WvB + (size_t)(nt * 16 + r) * Dd;
#pragma unroll
    for (int ks = 0; ks < 8; ++ks)
      c = wmma_bf(Ht[ks], b_from_bf16(wrow + ks * 32 + kb), c);
    const float bias = bv[nt * 16 + r];
    v8bf pk;
#pragma unroll
    for (int i = 0; i < 8; ++i) pk[i] = (__bf16)(c[i] + bias);
    *(v8bf*)(Vt + ((size_t)bkidx * Dd + nt * 16 + r) * Nn + nbase + hi8) = pk;
  }
}

// ---------------------------------------------------------------------------
// Kernel 2: one recurrence step. grid = B, block = 256 (8 waves).
// M dimension (L=8 slots) padded to 16 with zero rows so WMMA runs EXEC=all.
// ---------------------------------------------------------------------------
__global__ __launch_bounds__(256, 1) void slot_step_kernel(
    int kstep, float* __restrict__ slots, const __bf16* __restrict__ Kp,
    const __bf16* __restrict__ Vt, const __bf16* __restrict__ WqB,
    const __bf16* __restrict__ WihB, const __bf16* __restrict__ WhhB,
    const __bf16* __restrict__ W1B, const __bf16* __restrict__ W2B,
    const float* __restrict__ bq, const float* __restrict__ bih,
    const float* __restrict__ bhh, const float* __restrict__ b1,
    const float* __restrict__ b2, const float* __restrict__ lng,
    const float* __restrict__ lnb, float* __restrict__ Sout,
    float* __restrict__ Beta) {
  __shared__ float  SLOT[16][LDP];  // slots (rows 8..15 = 0)
  __shared__ float  QS[16][LDP];    // Q (scaled); reused as FFN hidden
  __shared__ float  SC[16][SCP];    // scores; first 16x256 floats reused as X
  __shared__ __bf16 AT[16][ATP];    // attn (bf16), pad rows = 0
  __shared__ float  UPD[16][LDP];
  __shared__ float  S2[16][LDP];    // total ~166 KB of 320 KB WGP LDS

  const int    b    = blockIdx.x;
  const int    tid  = threadIdx.x;
  const int    wave = tid >> 5;
  const int    lane = tid & 31;
  const int    r    = lane & 15;
  const int    kb   = (lane >> 4) * 8;
  const int    hi8  = (lane >> 4) * 8;
  const size_t bk   = (size_t)b * Kk + kstep;

  // ---- init: slots into LDS (zero-padded), zero attn pad rows ----
  for (int i = tid; i < 16 * Dd; i += 256) {
    int m = i >> 8, d = i & 255;
    SLOT[m][d] = (m < Ll) ? slots[((size_t)b * Ll + m) * Dd + d] : 0.f;
  }
  for (int i = tid; i < 8 * Nn; i += 256)
    AT[Ll + (i >> 10)][i & (Nn - 1)] = (__bf16)0.f;
  __syncthreads();

  // ---- Q = (SLOT @ Wq^T + bq) * 1/sqrt(D)  (scale folded here) ----
#pragma unroll 1
  for (int t = wave * 2; t < wave * 2 + 2; ++t) {
    v8f c = {};
    const __bf16* wrow = WqB + (size_t)(t * 16 + r) * Dd;
#pragma unroll
    for (int ks = 0; ks < 8; ++ks)
      c = wmma_bf(a_from_f32(&SLOT[r][ks * 32 + kb]),
                  b_from_bf16(wrow + ks * 32 + kb), c);
    const float bias = bq[t * 16 + r];
#pragma unroll
    for (int i = 0; i < 8; ++i)
      QS[i + hi8][t * 16 + r] = (c[i] + bias) * 0.0625f;
  }
  __syncthreads();

  // ---- scores (swapped): C[n][slot] = K[n][:] . Q[slot][:] ----
  // Each wave owns a 128-wide N slab; lane writes 8 consecutive n -> packed.
  {
    v16bf qb[8];
#pragma unroll
    for (int ks = 0; ks < 8; ++ks) qb[ks] = a_from_f32(&QS[r][ks * 32 + kb]);
    const int nb = wave * 128;
#pragma unroll 1
    for (int nt = 0; nt < 8; ++nt) {
      const int n0 = nb + nt * 16;
      v8f       c  = {};
      const __bf16* krow = Kp + (bk * Nn + n0 + r) * Dd;
#pragma unroll
      for (int ks = 0; ks < 8; ++ks)
        c = wmma_bf(b_from_bf16(krow + ks * 32 + kb), qb[ks], c);
      // lane: slot row = r, n = n0 + hi8 + {0..7}; pad slots r>=8 land in
      // unused SC rows.
      float4 s0 = {c[0], c[1], c[2], c[3]};
      float4 s1 = {c[4], c[5], c[6], c[7]};
      *(float4*)&SC[r][n0 + hi8]     = s0;
      *(float4*)&SC[r][n0 + hi8 + 4] = s1;
    }
  }
  __syncthreads();

  // ---- softmax over N per slot row; emit Beta (f32) + attn (bf16) ----
  if (wave < Ll) {
    const int row = wave;
    float     mx  = -3.4e38f;
    for (int j = lane; j < Nn; j += 32) mx = fmaxf(mx, SC[row][j]);
    mx = wave_max(mx);
    float s = 0.f;
    for (int j = lane; j < Nn; j += 32) {
      float e = __expf(SC[row][j] - mx);
      SC[row][j] = e;
      s += e;
    }
    s               = wave_sum(s);
    const float inv = 1.f / s;
    float* brow     = Beta + (bk * Ll + row) * Nn;
    for (int j = lane; j < Nn; j += 32) {
      float p = SC[row][j] * inv;
      brow[j] = p;
      AT[row][j] = (__bf16)p;
    }
  }
  __syncthreads();

  // ---- upd = attn @ V  (K = 1024, B-tiles from transposed Vt) ----
#pragma unroll 1
  for (int t = wave * 2; t < wave * 2 + 2; ++t) {
    v8f c = {};
    const __bf16* vrow = Vt + (bk * Dd + t * 16 + r) * Nn;
#pragma unroll 1
    for (int ks = 0; ks < 32; ++ks)
      c = wmma_bf(b_from_bf16(&AT[r][ks * 32 + kb]),
                  b_from_bf16(vrow + ks * 32 + kb), c);
#pragma unroll
    for (int i = 0; i < 8; ++i) UPD[i + hi8][t * 16 + r] = c[i];
  }
  __syncthreads();

  // ---- GRU cell: 6 fused WMMA accumulators per d-tile, gates in regs ----
#pragma unroll 1
  for (int t = wave * 2; t < wave * 2 + 2; ++t) {
    v8f       cir = {}, ciz = {}, cin = {}, chr2 = {}, chz = {}, chn = {};
    const int d   = t * 16 + r;
    const __bf16* wir = WihB + (size_t)d * Dd;
    const __bf16* wiz = WihB + (size_t)(Dd + d) * Dd;
    const __bf16* win = WihB + (size_t)(2 * Dd + d) * Dd;
    const __bf16* whr = WhhB + (size_t)d * Dd;
    const __bf16* whz = WhhB + (size_t)(Dd + d) * Dd;
    const __bf16* whn = WhhB + (size_t)(2 * Dd + d) * Dd;
#pragma unroll 1
    for (int ks = 0; ks < 8; ++ks) {
      const int o  = ks * 32 + kb;
      v16bf     au = a_from_f32(&UPD[r][o]);
      v16bf     as = a_from_f32(&SLOT[r][o]);
      cir  = wmma_bf(au, b_from_bf16(wir + o), cir);
      ciz  = wmma_bf(au, b_from_bf16(wiz + o), ciz);
      cin  = wmma_bf(au, b_from_bf16(win + o), cin);
      chr2 = wmma_bf(as, b_from_bf16(whr + o), chr2);
      chz  = wmma_bf(as, b_from_bf16(whz + o), chz);
      chn  = wmma_bf(as, b_from_bf16(whn + o), chn);
    }
    const float bir = bih[d], biz = bih[Dd + d], bin = bih[2 * Dd + d];
    const float bhr = bhh[d], bhz = bhh[Dd + d], bhn = bhh[2 * Dd + d];
#pragma unroll
    for (int i = 0; i < 8; ++i) {
      int   m  = i + hi8;
      float rg = sigm((cir[i] + bir) + (chr2[i] + bhr));
      float zg = sigm((ciz[i] + biz) + (chz[i] + bhz));
      float nn = tanh_fast((cin[i] + bin) + rg * (chn[i] + bhn));
      S2[m][d] = (1.f - zg) * nn + zg * SLOT[m][d];
    }
  }
  __syncthreads();

  // ---- FFN layer 1: relu(S2 @ W1^T + b1) -> QS (reuse) ----
#pragma unroll 1
  for (int t = wave * 2; t < wave * 2 + 2; ++t) {
    v8f c = {};
    const __bf16* wrow = W1B + (size_t)(t * 16 + r) * Dd;
#pragma unroll
    for (int ks = 0; ks < 8; ++ks)
      c = wmma_bf(a_from_f32(&S2[r][ks * 32 + kb]),
                  b_from_bf16(wrow + ks * 32 + kb), c);
    const float bias = b1[t * 16 + r];
#pragma unroll
    for (int i = 0; i < 8; ++i)
      QS[i + hi8][t * 16 + r] = fmaxf(c[i] + bias, 0.f);
  }
  __syncthreads();

  // ---- FFN layer 2 + residual: X = S2 + QS @ W2^T + b2 (X aliases SC) ----
  float* X = &SC[0][0];  // flat [16][Dd] view inside SC storage
#pragma unroll 1
  for (int t = wave * 2; t < wave * 2 + 2; ++t) {
    v8f c = {};
    const __bf16* wrow = W2B + (size_t)(t * 16 + r) * Dd;
#pragma unroll
    for (int ks = 0; ks < 8; ++ks)
      c = wmma_bf(a_from_f32(&QS[r][ks * 32 + kb]),
                  b_from_bf16(wrow + ks * 32 + kb), c);
    const int   d    = t * 16 + r;
    const float bias = b2[d];
#pragma unroll
    for (int i = 0; i < 8; ++i) {
      int m = i + hi8;
      X[m * Dd + d] = S2[m][d] + c[i] + bias;
    }
  }
  __syncthreads();

  // ---- LayerNorm + outputs (population variance, eps=1e-5) ----
  if (wave < Ll) {
    const int row = wave;
    float     s = 0.f, sq = 0.f;
    for (int j = lane; j < Dd; j += 32) {
      float x = X[row * Dd + j];
      s += x;
      sq += x * x;
    }
    s  = wave_sum(s);
    sq = wave_sum(sq);
    const float mean = s * (1.f / Dd);
    const float var  = sq * (1.f / Dd) - mean * mean;
    const float inv  = rsqrtf(var + 1e-5f);
    for (int j = lane; j < Dd; j += 32) {
      float y = (X[row * Dd + j] - mean) * inv * lng[j] + lnb[j];
      Sout[(bk * Ll + row) * Dd + j]         = y;  // S [B,K,L,D]
      slots[((size_t)b * Ll + row) * Dd + j] = y;  // carry to next step
    }
  }
}

// ---------------------------------------------------------------------------
extern "C" void kernel_launch(void* const* d_in, const int* in_sizes, int n_in,
                              void* d_out, int out_size, void* d_ws,
                              size_t ws_size, hipStream_t stream) {
  const float* H   = (const float*)d_in[0];
  const float* eps = (const float*)d_in[1];
  const float* mu  = (const float*)d_in[2];
  const float* ls  = (const float*)d_in[3];
  const float* Wq  = (const float*)d_in[4];
  const float* bq  = (const float*)d_in[5];
  const float* Wk  = (const float*)d_in[6];
  const float* bk  = (const float*)d_in[7];
  const float* Wv  = (const float*)d_in[8];
  const float* bv  = (const float*)d_in[9];
  const float* Wih = (const float*)d_in[10];
  const float* bih = (const float*)d_in[11];
  const float* Whh = (const float*)d_in[12];
  const float* bhh = (const float*)d_in[13];
  const float* W1  = (const float*)d_in[14];
  const float* b1  = (const float*)d_in[15];
  const float* W2  = (const float*)d_in[16];
  const float* b2  = (const float*)d_in[17];
  const float* lng = (const float*)d_in[18];
  const float* lnb = (const float*)d_in[19];

  char* p = (char*)d_ws;
  auto take = [&](size_t bytes) -> char* {
    char* q = p;
    p += (bytes + 255) & ~(size_t)255;
    return q;
  };
  __bf16* wqB   = (__bf16*)take((size_t)Dd * Dd * 2);
  __bf16* wkB   = (__bf16*)take((size_t)Dd * Dd * 2);
  __bf16* wvB   = (__bf16*)take((size_t)Dd * Dd * 2);
  __bf16* wihB  = (__bf16*)take((size_t)3 * Dd * Dd * 2);
  __bf16* whhB  = (__bf16*)take((size_t)3 * Dd * Dd * 2);
  __bf16* w1B   = (__bf16*)take((size_t)Dd * Dd * 2);
  __bf16* w2B   = (__bf16*)take((size_t)Dd * Dd * 2);
  float*  slots = (float*)take((size_t)Bb * Ll * Dd * 4);
  __bf16* Kp    = (__bf16*)take((size_t)Bb * Kk * Nn * Dd * 2);  // 128 MB
  __bf16* Vt    = (__bf16*)take((size_t)Bb * Kk * Nn * Dd * 2);  // 128 MB

  prep_kernel<<<768, 256, 0, stream>>>(eps, mu, ls, Wq, Wk, Wv, Wih, Whh, W1,
                                       W2, wqB, wkB, wvB, wihB, whhB, w1B, w2B,
                                       slots);
  kv_proj_kernel<<<Bb * Kk * 8, 256, 0, stream>>>(H, wkB, wvB, bk, bv, Kp, Vt);

  float* Sout = (float*)d_out;
  float* Beta = Sout + (size_t)Bb * Kk * Ll * Dd;
  for (int k = 0; k < Kk; ++k)
    slot_step_kernel<<<Bb, 256, 0, stream>>>(k, slots, Kp, Vt, wqB, wihB, whhB,
                                             w1B, w2B, bq, bih, bhh, b1, b2,
                                             lng, lnb, Sout, Beta);
}
